// DeepIterativeNetwork_33165737459875
// MI455X (gfx1250) — compile-verified, matrix operand order checked by reading, not run on hardware
//
#include <hip/hip_runtime.h>
#include <hip/hip_bf16.h>

#define NDEV 100000
#define DDEG 4
#define NBRK 200000
#define EDIM 128
#define TITER 3

typedef __attribute__((ext_vector_type(16))) _Float16 v16h;
typedef __attribute__((ext_vector_type(8)))  float    v8f;

// POD helper types (avoid HIP vector-type ctor issues inside unions)
struct __align__(16) B16 { unsigned u0, u1, u2, u3; };
struct __align__(16) F4  { float x, y, z, w; };
struct __align__(16) I4  { int x, y, z, w; };
struct __align__(8)  I2  { int x, y; };

union HFrag { v16h v; B16 q[2]; };
union FAcc  { v8f v; float f[8]; };

// ---- branch-free activations (native v_exp_f32 + v_rcp_f32, no exec churn) ----
__device__ __forceinline__ float fast_tanh(float x) {
    x = fminf(15.0f, fmaxf(-15.0f, x));
    const float e = __expf(2.0f * x);
    return (e - 1.0f) * __builtin_amdgcn_rcpf(e + 1.0f);
}
__device__ __forceinline__ float fast_sigm(float x) {
    x = fminf(30.0f, fmaxf(-30.0f, x));
    return __builtin_amdgcn_rcpf(1.0f + __expf(-x));
}

// ---- fragment loaders (CDNA5 WMMA 16x16x32 f16 layouts, ISA 7.12.2) ----

// A-matrix 16x32 f16: lane L (m = L&15, h = L>>4) holds K = 8h+{0..7} and 8h+16+{0..7}
__device__ __forceinline__ v16h load_frag_a(const _Float16* S, int ld, int mrow,
                                            int k0, int lane) {
    const int h = lane >> 4, m = lane & 15;
    HFrag f;
    const _Float16* p = S + (size_t)(mrow + m) * ld + k0 + 8 * h;
    f.q[0] = *(const B16*)p;
    f.q[1] = *(const B16*)(p + 16);
    return f.v;
}

// B-matrix 32x16 f16 with B[k][n] = W[n][k] (W row-major, row = out feature, ld = K):
// lane L holds column n = L&15; per-lane K chunks mirror the A layout -> contiguous in W row.
__device__ __forceinline__ v16h load_frag_b(const _Float16* W, int ldw, int nrow,
                                            int k0, int lane) {
    const int h = lane >> 4, c = lane & 15;
    HFrag f;
    const _Float16* p = W + (size_t)(nrow + c) * ldw + k0 + 8 * h;
    f.q[0] = *(const B16*)p;
    f.q[1] = *(const B16*)(p + 16);
    return f.v;
}

__device__ __forceinline__ v8f wmma_f16(v16h a, v16h b, v8f c) {
    return __builtin_amdgcn_wmma_f32_16x16x32_f16(false, a, false, b,
                                                  (short)0, c, false, false);
}

// ---------------------------------------------------------------------------
// Kernel: convert weights to f16 in workspace.
// wh layout (half index): W0h[0..16384) W3h[16384..32768) W5h[32768..49152) Wuh[49152..147456)
__global__ void prep_weights(const float* __restrict__ W0, const float* __restrict__ W3,
                             const float* __restrict__ W5, const float* __restrict__ Wu,
                             _Float16* __restrict__ wh) {
    int i = blockIdx.x * blockDim.x + threadIdx.x;
    if (i < 16384)        wh[i] = (_Float16)W0[i];
    else if (i < 32768)   wh[i] = (_Float16)W3[i - 16384];
    else if (i < 49152)   wh[i] = (_Float16)W5[i - 32768];
    else if (i < 147456)  wh[i] = (_Float16)Wu[i - 49152];
}

// Kernel: V_ws = broadcast(v0)
__global__ void init_v(const float* __restrict__ v0, float* __restrict__ Vws) {
    int i = blockIdx.x * blockDim.x + threadIdx.x;
    if (i < NDEV * EDIM) Vws[i] = v0[i & (EDIM - 1)];
}

// ---------------------------------------------------------------------------
// Precompute pe/be (f16): tanh-presums in LDS, then WMMA GEMMs vs W0h/W3h.
__global__ __launch_bounds__(256) void precompute_k(
    const float* __restrict__ prot, const float* __restrict__ bstate,
    const int* __restrict__ dbi_,
    const _Float16* __restrict__ W0h, const _Float16* __restrict__ W3h,
    const float* __restrict__ b0, const float* __restrict__ b3,
    const float* __restrict__ w1, const float* __restrict__ b1,
    const float* __restrict__ w2, const float* __restrict__ b2,
    const float* __restrict__ w4, const float* __restrict__ b4,
    _Float16* __restrict__ peh, _Float16* __restrict__ beh) {
    __shared__ float sScal[128][8];
    __shared__ __align__(16) _Float16 sBE[128][136];
    __shared__ __align__(16) _Float16 sPE[128][136];

    const int tid = threadIdx.x;
    const int base = blockIdx.x * 128;
    const bool fullblk = (base + 128 <= NDEV);

    // phase 0: per-row scalars
    if (tid < 128) {
        const int gr = base + tid;
        float c0 = 0.f, c1 = 0.f, c2 = 0.f, c3 = 0.f, p0 = 0.f, p1 = 0.f, p2 = 0.f;
        if (gr < NDEV) {
            const I4 db = ((const I4*)dbi_)[gr];
            c0 = bstate[db.x]; c1 = bstate[db.y]; c2 = bstate[db.z]; c3 = bstate[db.w];
            p0 = prot[gr * 3 + 0]; p1 = prot[gr * 3 + 1]; p2 = prot[gr * 3 + 2];
        }
        sScal[tid][0] = c0; sScal[tid][1] = c1; sScal[tid][2] = c2; sScal[tid][3] = c3;
        sScal[tid][4] = c0 + c1 + c2 + c3;
        sScal[tid][5] = p0; sScal[tid][6] = p1; sScal[tid][7] = p2;
    }
    __syncthreads();

    // phase 1: element-wise presums -> LDS f16 A-matrices (branch-free tanh)
    {
        const int e = tid & 127;
        const float w4e = w4[e], b4e = b4[e];
        const float w1e = w1[e], b1e = b1[e];
        const float w2e = w2[e], b2e = b2[e];
        for (int idx = tid; idx < 128 * 128; idx += 256) {
            const int r = idx >> 7;
            const float c0 = sScal[r][0], c1 = sScal[r][1], c2 = sScal[r][2],
                        c3 = sScal[r][3], cs = sScal[r][4];
            const float p0 = sScal[r][5], p1 = sScal[r][6], p2 = sScal[r][7];
            float bes = fast_tanh(c0 * w4e + b4e) + fast_tanh(c1 * w4e + b4e) +
                        fast_tanh(c2 * w4e + b4e) + fast_tanh(c3 * w4e + b4e);
            float pes = fast_tanh(p0 * w1e + b1e) + fast_tanh(p1 * w1e + b1e) +
                        fast_tanh(p2 * w1e + b1e) + 3.0f * fast_tanh(cs * w2e + b2e);
            sBE[r][e] = (_Float16)bes;
            sPE[r][e] = (_Float16)pes;
        }
    }
    __syncthreads();

    // phase 2: WMMA GEMMs  be = tanh(sBE @ W3^T + b3), pe = tanh(sPE @ W0^T + b0)
    const int lane = tid & 31, wave = tid >> 5;
    const int mrow = wave * 16, h = lane >> 4, cc = lane & 15;
    for (int pass = 0; pass < 2; ++pass) {
        const _Float16* S  = pass ? &sPE[0][0] : &sBE[0][0];
        const _Float16* Wh = pass ? W0h : W3h;
        const float* bias  = pass ? b0 : b3;
        _Float16* Out      = pass ? peh : beh;
        for (int nsub = 0; nsub < 8; ++nsub) {
            const int nbase = nsub * 16;
            FAcc acc;
#pragma unroll
            for (int i = 0; i < 8; ++i) acc.f[i] = 0.f;
#pragma unroll
            for (int ko = 0; ko < 4; ++ko) {
                v16h a = load_frag_a(S, 136, mrow, ko * 32, lane);
                v16h b = load_frag_b(Wh, 128, nbase, ko * 32, lane);
                acc.v = wmma_f16(a, b, acc.v);
            }
            const int n = nbase + cc;
            const float bn = bias[n];
            if (fullblk) {
#pragma unroll
                for (int r = 0; r < 8; ++r) {
                    const int gr = base + mrow + r + 8 * h;
                    Out[(size_t)gr * EDIM + n] = (_Float16)fast_tanh(acc.f[r] + bn);
                }
            } else {
#pragma unroll
                for (int r = 0; r < 8; ++r) {
                    const int gr = base + mrow + r + 8 * h;
                    if (gr < NDEV)
                        Out[(size_t)gr * EDIM + n] = (_Float16)fast_tanh(acc.f[r] + bn);
                }
            }
        }
    }
}

// ---------------------------------------------------------------------------
// One message-passing iteration: gather -> WMMA(W5)+info -> WMMA(Wu,K=256)+GRU update.
__global__ __launch_bounds__(256) void iterate_k(
    const float* __restrict__ Vin, float* __restrict__ Vout,
    const _Float16* __restrict__ peh, const _Float16* __restrict__ beh,
    const _Float16* __restrict__ W5h, const _Float16* __restrict__ Wuh,
    const float* __restrict__ b5, const float* __restrict__ wc,
    const float* __restrict__ bcp, const float* __restrict__ bu,
    const int* __restrict__ dbi_, const int* __restrict__ pairs_) {
    __shared__ __align__(16) _Float16 sNS[128][136];   // sum of 4 neighbor rows
    __shared__ __align__(16) _Float16 sVI[128][136];   // own V row
    __shared__ __align__(16) _Float16 sINFO[128][136]; // fused conv output

    const int tid = threadIdx.x;
    const int base = blockIdx.x * 128;
    const bool fullblk = (base + 128 <= NDEV);

    // phase 0: neighbor gather + own-row stage (2 threads per row, 64 cols each)
    {
        const int r = tid >> 1, sub = tid & 1;
        const int gr = base + r;
        const int i4base = sub * 16; // float4 index within row (E/4 = 32 total)
        if (fullblk || gr < NDEV) {
            const I4 db = ((const I4*)dbi_)[gr];
            const I2* pr2 = (const I2*)pairs_;
            const I2 q0 = pr2[db.x], q1 = pr2[db.y], q2 = pr2[db.z], q3 = pr2[db.w];
            const int n0 = (q0.x == gr) ? q0.y : q0.x;
            const int n1 = (q1.x == gr) ? q1.y : q1.x;
            const int n2 = (q2.x == gr) ? q2.y : q2.x;
            const int n3 = (q3.x == gr) ? q3.y : q3.x;
            const F4* v4 = (const F4*)Vin;
            const F4* r0 = v4 + (size_t)n0 * 32;
            const F4* r1 = v4 + (size_t)n1 * 32;
            const F4* r2 = v4 + (size_t)n2 * 32;
            const F4* r3 = v4 + (size_t)n3 * 32;
            const F4* ro = v4 + (size_t)gr * 32;
#pragma unroll
            for (int j = 0; j < 16; ++j) {
                const int i4 = i4base + j;
                const F4 a = r0[i4], b = r1[i4], c = r2[i4], d = r3[i4], o = ro[i4];
                const int e = i4 * 4;
                sNS[r][e + 0] = (_Float16)(a.x + b.x + c.x + d.x);
                sNS[r][e + 1] = (_Float16)(a.y + b.y + c.y + d.y);
                sNS[r][e + 2] = (_Float16)(a.z + b.z + c.z + d.z);
                sNS[r][e + 3] = (_Float16)(a.w + b.w + c.w + d.w);
                sVI[r][e + 0] = (_Float16)o.x;
                sVI[r][e + 1] = (_Float16)o.y;
                sVI[r][e + 2] = (_Float16)o.z;
                sVI[r][e + 3] = (_Float16)o.w;
            }
        } else {
#pragma unroll
            for (int j = 0; j < 16; ++j) {
                const int e = (i4base + j) * 4;
#pragma unroll
                for (int u = 0; u < 4; ++u) {
                    sNS[r][e + u] = (_Float16)0.f;
                    sVI[r][e + u] = (_Float16)0.f;
                }
            }
        }
    }
    __syncthreads();

    const int lane = tid & 31, wave = tid >> 5;
    const int mrow = wave * 16, h = lane >> 4, cc = lane & 15;
    const float wc0 = wc[0], wc1 = wc[1], wc2 = wc[2], bcv = bcp[0];

    // phase 1: ne = tanh(sNS @ W5^T + b5); info = tanh(wc·[pe,be,ne] + bc) -> LDS
    for (int nsub = 0; nsub < 8; ++nsub) {
        const int nbase = nsub * 16;
        FAcc acc;
#pragma unroll
        for (int i = 0; i < 8; ++i) acc.f[i] = 0.f;
#pragma unroll
        for (int ko = 0; ko < 4; ++ko) {
            v16h a = load_frag_a(&sNS[0][0], 136, mrow, ko * 32, lane);
            v16h b = load_frag_b(W5h, 128, nbase, ko * 32, lane);
            acc.v = wmma_f16(a, b, acc.v);
        }
        const int n = nbase + cc;
        const float b5n = b5[n];
        if (fullblk) {
#pragma unroll
            for (int r = 0; r < 8; ++r) {
                const int m = mrow + r + 8 * h;
                const size_t gi = (size_t)(base + m) * EDIM + n;
                const float ne = fast_tanh(acc.f[r] + b5n);
                const float info =
                    fast_tanh(wc0 * (float)peh[gi] + wc1 * (float)beh[gi] + wc2 * ne + bcv);
                sINFO[m][n] = (_Float16)info;
            }
        } else {
#pragma unroll
            for (int r = 0; r < 8; ++r) {
                const int m = mrow + r + 8 * h;
                const int gr = base + m;
                const float ne = fast_tanh(acc.f[r] + b5n);
                float pev = 0.f, bev = 0.f;
                if (gr < NDEV) {
                    pev = (float)peh[(size_t)gr * EDIM + n];
                    bev = (float)beh[(size_t)gr * EDIM + n];
                }
                sINFO[m][n] = (_Float16)fast_tanh(wc0 * pev + wc1 * bev + wc2 * ne + bcv);
            }
        }
    }
    __syncthreads();

    // phase 2: gates = [V|info] @ Wu^T (K=256); keep/upd/new computed as 3 parallel
    // accumulators per 16x16 tile so the GRU update fuses with no 3E staging.
    for (int ns = 0; ns < 8; ++ns) {
        const int nb = ns * 16;
        FAcc aK, aU, aN;
#pragma unroll
        for (int i = 0; i < 8; ++i) { aK.f[i] = 0.f; aU.f[i] = 0.f; aN.f[i] = 0.f; }
#pragma unroll
        for (int kk = 0; kk < 8; ++kk) {
            const int k0 = kk * 32;
            v16h a = (k0 < 128)
                         ? load_frag_a(&sVI[0][0], 136, mrow, k0, lane)
                         : load_frag_a(&sINFO[0][0], 136, mrow, k0 - 128, lane);
            v16h bK = load_frag_b(Wuh, 256, nb, k0, lane);
            v16h bU = load_frag_b(Wuh, 256, 128 + nb, k0, lane);
            v16h bN = load_frag_b(Wuh, 256, 256 + nb, k0, lane);
            aK.v = wmma_f16(a, bK, aK.v);
            aU.v = wmma_f16(a, bU, aU.v);
            aN.v = wmma_f16(a, bN, aN.v);
        }
        const int e = nb + cc;
        const float buK = bu[e], buU = bu[128 + e], buN = bu[256 + e];
        if (fullblk) {
#pragma unroll
            for (int r = 0; r < 8; ++r) {
                const int m = mrow + r + 8 * h;
                const float keep = aK.f[r] + buK;
                const float upd  = aU.f[r] + buU;
                const float nw   = aN.f[r] + buN;
                const float vold = (float)sVI[m][e];
                const float vnew =
                    fast_tanh(vold * fast_sigm(keep) + fast_sigm(upd) * fast_tanh(nw));
                Vout[(size_t)(base + m) * EDIM + e] = vnew;
            }
        } else {
#pragma unroll
            for (int r = 0; r < 8; ++r) {
                const int m = mrow + r + 8 * h;
                const int gr = base + m;
                const float keep = aK.f[r] + buK;
                const float upd  = aU.f[r] + buU;
                const float nw   = aN.f[r] + buN;
                const float vold = (float)sVI[m][e];
                const float vnew =
                    fast_tanh(vold * fast_sigm(keep) + fast_sigm(upd) * fast_tanh(nw));
                if (gr < NDEV) Vout[(size_t)gr * EDIM + e] = vnew;
            }
        }
    }
}

// ---------------------------------------------------------------------------
__global__ void zero_colsum(float* __restrict__ colsum) {
    colsum[threadIdx.x] = 0.f;
}

__global__ __launch_bounds__(256) void colsum_partial(const float* __restrict__ V,
                                                      float* __restrict__ colsum) {
    const int tid = threadIdx.x;
    const int col = tid & 127, half = tid >> 7;
    float acc = 0.f;
    for (int r = blockIdx.x * 2 + half; r < NDEV; r += gridDim.x * 2)
        acc += V[(size_t)r * EDIM + col];
    __shared__ float sc[256];
    sc[tid] = acc;
    __syncthreads();
    if (tid < 128) atomicAdd(&colsum[col], sc[tid] + sc[tid + 128]);
}

__global__ void grid_emb_k(const float* __restrict__ colsum, const float* __restrict__ Wg,
                           const float* __restrict__ bg, float* __restrict__ out) {
    const int n = threadIdx.x;
    float a = bg[n];
    for (int k = 0; k < EDIM; ++k) a += colsum[k] * Wg[n * EDIM + k];
    out[n] = a;
}

// ---------------------------------------------------------------------------
extern "C" void kernel_launch(void* const* d_in, const int* in_sizes, int n_in,
                              void* d_out, int out_size, void* d_ws, size_t ws_size,
                              hipStream_t stream) {
    (void)in_sizes; (void)n_in; (void)out_size; (void)ws_size;

    const float* prot   = (const float*)d_in[0];
    const float* bstate = (const float*)d_in[1];
    const float* v0     = (const float*)d_in[2];
    const float* W0     = (const float*)d_in[3];
    const float* b0     = (const float*)d_in[4];
    const float* w1     = (const float*)d_in[5];
    const float* b1     = (const float*)d_in[6];
    const float* w2     = (const float*)d_in[7];
    const float* b2     = (const float*)d_in[8];
    const float* W3     = (const float*)d_in[9];
    const float* b3     = (const float*)d_in[10];
    const float* w4     = (const float*)d_in[11];
    const float* b4     = (const float*)d_in[12];
    const float* W5     = (const float*)d_in[13];
    const float* b5     = (const float*)d_in[14];
    const float* wc     = (const float*)d_in[15];
    const float* bc     = (const float*)d_in[16];
    const float* Wu     = (const float*)d_in[17];
    const float* bu     = (const float*)d_in[18];
    const float* Wg     = (const float*)d_in[19];
    const float* bg     = (const float*)d_in[20];
    const int*   dbi    = (const int*)d_in[21];
    const int*   pairs  = (const int*)d_in[22];

    char* ws = (char*)d_ws;
    _Float16* wh  = (_Float16*)ws;           // 147456 halfs of weights
    _Float16* W0h = wh;
    _Float16* W3h = wh + 16384;
    _Float16* W5h = wh + 32768;
    _Float16* Wuh = wh + 49152;
    float*    colsum = (float*)(ws + 294912);               // 128 floats
    _Float16* peh    = (_Float16*)(ws + 295424);            // N*E halfs
    _Float16* beh    = (_Float16*)(ws + 295424 + 25600000); // N*E halfs
    float*    Vws    = (float*)(ws + 51495424);             // N*E floats

    float* Vout = (float*)d_out;
    float* gout = Vout + (size_t)NDEV * EDIM;

    const int GB = (NDEV + 127) / 128; // 782

    prep_weights<<<(147456 + 255) / 256, 256, 0, stream>>>(W0, W3, W5, Wu, wh);
    init_v<<<(NDEV * EDIM + 255) / 256, 256, 0, stream>>>(v0, Vws);
    precompute_k<<<GB, 256, 0, stream>>>(prot, bstate, dbi, W0h, W3h, b0, b3,
                                         w1, b1, w2, b2, w4, b4, peh, beh);

    // T = 3, ping-pong so the final V lands in d_out
    iterate_k<<<GB, 256, 0, stream>>>(Vws, Vout, peh, beh, W5h, Wuh, b5, wc, bc, bu, dbi, pairs);
    iterate_k<<<GB, 256, 0, stream>>>(Vout, Vws, peh, beh, W5h, Wuh, b5, wc, bc, bu, dbi, pairs);
    iterate_k<<<GB, 256, 0, stream>>>(Vws, Vout, peh, beh, W5h, Wuh, b5, wc, bc, bu, dbi, pairs);

    zero_colsum<<<1, 128, 0, stream>>>(colsum);
    colsum_partial<<<256, 256, 0, stream>>>(Vout, colsum);
    grid_emb_k<<<1, 128, 0, stream>>>(colsum, Wg, bg, gout);
}
